// LSTMController_35270271435219
// MI455X (gfx1250) — matrix operand order repaired and checked by reading, not compiled, not run on hardware
//
#include <hip/hip_runtime.h>
#include <hip/hip_bf16.h>

typedef __attribute__((ext_vector_type(16))) _Float16 v16h;
typedef __attribute__((ext_vector_type(8)))  float    v8f;
typedef __attribute__((ext_vector_type(4)))  unsigned u32x4;
typedef __attribute__((ext_vector_type(8)))  unsigned u32x8;

#define LL 2048
#define HH 1024
#define HALFD 512
#define GD 2048        // 4*HALF
#define BB 32
#define KK 32
#define SS 64
#define VV 50257
#define SCAN_BLOCKS 64
#define KSTEPS 32      // HH/32
#define TROW 40u       // padded LDS row stride in halves (64B payload + 16B pad)
#define TBUF 10240u    // 128*40*2 bytes per tile buffer

static __device__ __forceinline__ v8f v8f_zero() {
  v8f z = {0.f,0.f,0.f,0.f,0.f,0.f,0.f,0.f};
  return z;
}

static __device__ __forceinline__ float sigmoidf_(float x) {
  return 1.0f / (1.0f + __expf(-x));
}

// ---- WMMA fragment helpers (wave32, 16x16x32 f16, contiguous k = 16*hi + i in both A and B) ----
static __device__ __forceinline__ v16h load_a_f32(const float* __restrict__ row, int k0, int hi) {
  v16h a;
#pragma unroll
  for (int i = 0; i < 8; ++i) a[i]     = (_Float16)row[k0 + hi*16 + i];
#pragma unroll
  for (int i = 0; i < 8; ++i) a[8 + i] = (_Float16)row[k0 + hi*16 + 8 + i];
  return a;
}

static __device__ __forceinline__ v16h load_bT_f32(const float* __restrict__ wrow, int k0, int hi) {
  v16h b;
#pragma unroll
  for (int i = 0; i < 16; ++i) b[i] = (_Float16)wrow[k0 + hi*16 + i];
  return b;
}

static __device__ __forceinline__ v16h load_bT_f16(const _Float16* __restrict__ wrow, int k0, int hi) {
  v16h b;
#pragma unroll
  for (int i = 0; i < 16; ++i) b[i] = wrow[k0 + hi*16 + i];
  return b;
}

// ------------------------------------------------------------------
// Kernel 0: f32 -> f16 conversion (W_hh, W_ih, embed table)
// ------------------------------------------------------------------
__global__ void cvt_f16_kernel(const float* __restrict__ src, _Float16* __restrict__ dst, int n) {
  int i = blockIdx.x * 256 + threadIdx.x;
  if (i < n) dst[i] = (_Float16)src[i];
}

// ------------------------------------------------------------------
// Kernel 1: x_g = gather(emb16) @ W_ih16.T + (b_ih + b_hh), stored f16.
// Block = 256 thr (8 waves), tile 128 rows x 128 cols, double-buffered:
//  - B tile (128 W_ih rows x 32 halves) staged by TDM tensor_load_to_lds
//    with LDS padding (pad 4 DWORDs every 16 DWORDs -> 80B row stride),
//    pipelined with s_wait_tensorcnt 1.
//  - A tile (128 gathered embed rows x 32 halves) staged by per-lane
//    global_load_async_to_lds_b128, pipelined with s_wait_asynccnt 2.
// Each wave computes 32x64 of output (2x4 wmma tiles).
// ------------------------------------------------------------------
__global__ __launch_bounds__(256)
void xg_gemm_kernel(const int* __restrict__ seq,
                    const _Float16* __restrict__ embed16,
                    const _Float16* __restrict__ wih16,
                    const float* __restrict__ bih,
                    const float* __restrict__ bhh,
                    _Float16* __restrict__ xg) {
  __shared__ _Float16 aT[2][128 * TROW];
  __shared__ _Float16 bT[2][128 * TROW];

  const int wave = threadIdx.x >> 5, lane = threadIdx.x & 31;
  const int m = lane & 15, hi = lane >> 4;
  const int wm = wave >> 1;          // 0..3 -> rows wm*32
  const int wn = wave & 1;           // 0..1 -> cols wn*64
  const int row0 = blockIdx.x * 128;
  const int n0 = blockIdx.y * 128;

  // A-tile async gather: 512 16B chunks (128 rows x 4 chunks) over 256 thr
  unsigned rowoff[2];
#pragma unroll
  for (int i = 0; i < 2; ++i)
    rowoff[i] = (unsigned)seq[row0 + (threadIdx.x >> 2) + i * 64] * (unsigned)(HH * 2);
  const unsigned ch = (threadIdx.x & 3u) * 16u;
  const unsigned arow_l = ((unsigned)(threadIdx.x >> 2)) * (TROW * 2u) + ch;

  const unsigned aOff = __builtin_amdgcn_readfirstlane((unsigned)(size_t)&aT[0][0]);
  const unsigned bOff = __builtin_amdgcn_readfirstlane((unsigned)(size_t)&bT[0][0]);

  v8f acc[2][4];
#pragma unroll
  for (int mi = 0; mi < 2; ++mi)
#pragma unroll
    for (int ni = 0; ni < 4; ++ni) acc[mi][ni] = v8f_zero();

  auto issue_tile = [&](int it) {
    const unsigned nb = (unsigned)(it & 1);
    if (wave == 0) {
      // TDM: 2D f16 tile, tile_dim0=32 elems, tile_dim1=128 rows,
      // dim0_stride=1024, LDS pad: 4 DWORDs after every 16 DWORDs.
      unsigned long long gaddr =
          (unsigned long long)(size_t)(wih16 + (size_t)n0 * HH + it * 32);
      u32x4 g0;
      g0[0] = 1u;                                       // count=1, no gather
      g0[1] = bOff + nb * TBUF;                         // lds_addr
      g0[2] = (unsigned)gaddr;                          // global_addr[31:0]
      g0[3] = (unsigned)(gaddr >> 32) | 0x80000000u;    // addr[56:32] | type=2
      u32x8 g1;
      g1[0] = 0x06D10000u;   // data_size=2B | pad_enable | interval=3 | amount=3
      g1[1] = ((unsigned)HH) << 16;                     // tensor_dim0 = 1024
      g1[2] = ((unsigned)GD) << 16;                     // tensor_dim1 = 2048
      g1[3] = 32u << 16;                                // tile_dim0 = 32
      g1[4] = 128u;                                     // tile_dim1 = 128
      g1[5] = (unsigned)HH;                             // tensor_dim0_stride
      g1[6] = 0u;
      g1[7] = 0u;
      asm volatile("tensor_load_to_lds %0, %1" :: "s"(g0), "s"(g1) : "memory");
    }
#pragma unroll
    for (int i = 0; i < 2; ++i) {
      unsigned la = aOff + nb * TBUF + arow_l + (unsigned)i * (64u * TROW * 2u);
      unsigned go = rowoff[i] + (unsigned)(it * 64) + ch;   // it*32 halves * 2B
      asm volatile("global_load_async_to_lds_b128 %0, %1, %2"
                   :: "v"(la), "v"(go), "s"(embed16) : "memory");
    }
  };

  issue_tile(0);
  for (int it = 0; it < KSTEPS; ++it) {
    const int cur = it & 1;
    if (it + 1 < KSTEPS) {
      issue_tile(it + 1);
      asm volatile("s_wait_asynccnt 0x2" ::: "memory");
      if (wave == 0) asm volatile("s_wait_tensorcnt 0x1" ::: "memory");
    } else {
      asm volatile("s_wait_asynccnt 0x0" ::: "memory");
      if (wave == 0) asm volatile("s_wait_tensorcnt 0x0" ::: "memory");
    }
    __syncthreads();   // tile `it` complete & visible to all waves

    const _Float16* aB = &aT[cur][0];
    const _Float16* bB = &bT[cur][0];
    v16h afr[2];
#pragma unroll
    for (int mi = 0; mi < 2; ++mi) {
      const _Float16* ar = aB + (wm * 32 + mi * 16 + m) * TROW;
      v16h a;
#pragma unroll
      for (int i = 0; i < 8; ++i) a[i]     = ar[hi * 16 + i];
#pragma unroll
      for (int i = 0; i < 8; ++i) a[8 + i] = ar[hi * 16 + 8 + i];
      afr[mi] = a;
    }
#pragma unroll
    for (int ni = 0; ni < 4; ++ni) {
      const _Float16* br = bB + (wn * 64 + ni * 16 + m) * TROW;
      v16h b;
#pragma unroll
      for (int i = 0; i < 16; ++i) b[i] = br[hi * 16 + i];
#pragma unroll
      for (int mi = 0; mi < 2; ++mi)
        acc[mi][ni] = __builtin_amdgcn_wmma_f32_16x16x32_f16(false, afr[mi], false, b,
                                                             (short)0, acc[mi][ni], false, false);
    }
    __syncthreads();   // reads of buffer `cur` done before it is refilled
  }

#pragma unroll
  for (int mi = 0; mi < 2; ++mi)
#pragma unroll
    for (int ni = 0; ni < 4; ++ni)
#pragma unroll
      for (int i = 0; i < 8; ++i) {
        int rr = row0 + wm * 32 + mi * 16 + i + hi * 8;
        int n  = n0 + wn * 64 + ni * 16 + m;
        xg[(size_t)rr * GD + n] = (_Float16)(acc[mi][ni][i] + bih[n] + bhh[n]);
      }
}

// ------------------------------------------------------------------
// Grid-wide two-phase barrier (counter + generation) in global scratch
// ------------------------------------------------------------------
static __device__ __forceinline__ void grid_sync(unsigned* bar, unsigned nblk) {
  __syncthreads();
  if (threadIdx.x == 0) {
    volatile unsigned* gen = bar + 16;
    unsigned my_gen = *gen;
    __threadfence();
    unsigned arrived = atomicAdd(bar, 1u);
    if (arrived == nblk - 1) {
      atomicExch(bar, 0u);
      __threadfence();
      atomicAdd((unsigned*)(bar + 16), 1u);
    } else {
      while (*gen == my_gen) { __builtin_amdgcn_s_sleep(2); }
    }
    __threadfence();
  }
  __syncthreads();
}

// ------------------------------------------------------------------
// Kernel 2: persistent LSTM scan. 64 blocks x 128 thr = 256 waves =
// exactly the 2 (M) x 128 (N) tiles of the per-step (32x512)x(512x2048)
// GEMM. Per step: WMMA GEMM -> sync -> elementwise c/h -> sync -> gate.
// ------------------------------------------------------------------
__global__ void lstm_scan_kernel(const _Float16* __restrict__ xg,
                                 const _Float16* __restrict__ whh16,
                                 const float* __restrict__ Wg,
                                 const float* __restrict__ bg,
                                 float* __restrict__ h,
                                 float* __restrict__ c,
                                 float* __restrict__ gbuf,
                                 float* __restrict__ gate,
                                 unsigned* __restrict__ bar) {
  __shared__ float red[128];
  const int wave = threadIdx.x >> 5, lane = threadIdx.x & 31;
  const int wg = blockIdx.x * 4 + wave;    // 0..255
  const int mtile = wg & 1;                // batch tile (b 0-15 / 16-31)
  const int ntile = wg >> 1;               // 0..127 over GD
  const int m = lane & 15, hi = lane >> 4;
  const int tidg = blockIdx.x * 128 + threadIdx.x;

  for (int t = 0; t < LL; ++t) {
    // ---- phase 1: g = x_g[:,t,:] + h @ W_hh.T (one 16x16 tile per wave)
    v8f acc;
#pragma unroll
    for (int i = 0; i < 8; ++i) {
      int b = mtile*16 + i + hi*8;
      int n = ntile*16 + m;
      acc[i] = (float)xg[((size_t)b * LL + t) * GD + n];
    }
    for (int k0 = 0; k0 < HALFD; k0 += 32) {
      const float* hrow = h + (size_t)(mtile*16 + m) * HALFD;
      v16h a = load_a_f32(hrow, k0, hi);
      const _Float16* wrow = whh16 + (size_t)(ntile*16 + m) * HALFD;
      v16h b = load_bT_f16(wrow, k0, hi);
      acc = __builtin_amdgcn_wmma_f32_16x16x32_f16(false, a, false, b,
                                                   (short)0, acc, false, false);
    }
#pragma unroll
    for (int i = 0; i < 8; ++i) {
      int b = mtile*16 + i + hi*8;
      int n = ntile*16 + m;
      gbuf[b * GD + n] = acc[i];
    }

    grid_sync(bar, SCAN_BLOCKS);

    // ---- phase 2: elementwise LSTM cell (16384 elems over 8192 threads)
    for (int e = tidg; e < BB * HALFD; e += SCAN_BLOCKS * 128) {
      int b = e >> 9, j = e & 511;
      float gi = gbuf[b * GD + j];
      float gf = gbuf[b * GD + 512 + j];
      float gg = gbuf[b * GD + 1024 + j];
      float go = gbuf[b * GD + 1536 + j];
      float cc = sigmoidf_(gf) * c[b * HALFD + j] + sigmoidf_(gi) * tanhf(gg);
      c[b * HALFD + j] = cc;
      h[b * HALFD + j] = sigmoidf_(go) * tanhf(cc);
    }

    grid_sync(bar, SCAN_BLOCKS);

    // ---- phase 3: gate[b][t] = sigmoid(h[b]·Wg + bg), blocks 0..31
    if (blockIdx.x < BB) {
      int b = blockIdx.x;
      float s = 0.f;
      for (int j = threadIdx.x; j < HALFD; j += 128) s += h[b * HALFD + j] * Wg[j];
      red[threadIdx.x] = s;
      __syncthreads();
      for (int o = 64; o > 0; o >>= 1) {
        if (threadIdx.x < o) red[threadIdx.x] += red[threadIdx.x + o];
        __syncthreads();
      }
      if (threadIdx.x == 0) gate[b * LL + t] = sigmoidf_(red[0] + bg[0]);
      __syncthreads();
    }
  }
}

// ------------------------------------------------------------------
// Kernel 3: gumbel perturb + top-K(32) + softmax weights. 1 block/batch.
// ------------------------------------------------------------------
__global__ void topk_kernel(const float* __restrict__ gate,
                            const float* __restrict__ u,
                            int* __restrict__ topidx,
                            float* __restrict__ topw) {
  __shared__ float vals[LL];
  __shared__ float rv[256];
  __shared__ int   ri[256];
  __shared__ float tv[KK];
  __shared__ int   ti[KK];
  const int b = blockIdx.x;

  for (int l = threadIdx.x; l < LL; l += 256) {
    float uu = u[b * LL + l];
    uu = fminf(fmaxf(uu, 1e-10f), 1.0f);
    float gum = -__logf(-__logf(uu));
    vals[l] = gate[b * LL + l] + 0.1f * gum;
  }
  __syncthreads();

  for (int k = 0; k < KK; ++k) {
    float best = -1e30f; int bi = 0x7fffffff;
    for (int l = threadIdx.x; l < LL; l += 256) {
      float v = vals[l];
      if (v > best || (v == best && l < bi)) { best = v; bi = l; }
    }
    rv[threadIdx.x] = best; ri[threadIdx.x] = bi;
    __syncthreads();
    for (int o = 128; o > 0; o >>= 1) {
      if (threadIdx.x < o) {
        float v2 = rv[threadIdx.x + o]; int i2 = ri[threadIdx.x + o];
        if (v2 > rv[threadIdx.x] || (v2 == rv[threadIdx.x] && i2 < ri[threadIdx.x])) {
          rv[threadIdx.x] = v2; ri[threadIdx.x] = i2;
        }
      }
      __syncthreads();
    }
    if (threadIdx.x == 0) { tv[k] = rv[0]; ti[k] = ri[0]; vals[ri[0]] = -1e30f; }
    __syncthreads();
  }

  if (threadIdx.x < KK) {
    float mx = -1e30f;
    for (int k = 0; k < KK; ++k) mx = fmaxf(mx, tv[k]);
    float sum = 0.f;
    for (int k = 0; k < KK; ++k) sum += __expf(tv[k] - mx);
    int k = threadIdx.x;
    topw[b * KK + k]   = __expf(tv[k] - mx) / sum;
    topidx[b * KK + k] = ti[k];
  }
}

// ------------------------------------------------------------------
// Kernel 4: memory[b][s] = (emb_row @ Wv.T + bv) * w[b][s], top-K rows only.
// M = B*K = 1024 gathered rows; grid (64, 4), block 128 (4 waves).
// ------------------------------------------------------------------
__global__ void mem_gemm_kernel(const int* __restrict__ seq,
                                const float* __restrict__ embed,
                                const float* __restrict__ Wv,
                                const float* __restrict__ bv,
                                const int* __restrict__ topidx,
                                const float* __restrict__ topw,
                                float* __restrict__ memry) {
  const int wave = threadIdx.x >> 5, lane = threadIdx.x & 31;
  const int m = lane & 15, hi = lane >> 4;
  const int row0 = blockIdx.x * 16;               // rk = b*K + s
  const int n0 = (blockIdx.y * 4 + wave) * 64;

  const int rk = row0 + m;
  const int b = rk >> 5, s = rk & 31;
  const int l = topidx[b * KK + s];
  const float* arow = embed + (size_t)seq[b * LL + l] * HH;

  v8f acc[4];
#pragma unroll
  for (int nt = 0; nt < 4; ++nt) acc[nt] = v8f_zero();

  for (int k0 = 0; k0 < HH; k0 += 32) {
    v16h a = load_a_f32(arow, k0, hi);
#pragma unroll
    for (int nt = 0; nt < 4; ++nt) {
      const float* wrow = Wv + (size_t)(n0 + nt*16 + m) * HH;
      v16h bf = load_bT_f32(wrow, k0, hi);
      acc[nt] = __builtin_amdgcn_wmma_f32_16x16x32_f16(false, a, false, bf,
                                                       (short)0, acc[nt], false, false);
    }
  }
#pragma unroll
  for (int nt = 0; nt < 4; ++nt) {
#pragma unroll
    for (int i = 0; i < 8; ++i) {
      int rr = row0 + i + hi*8;
      int bb = rr >> 5, ss = rr & 31;
      int n = n0 + nt*16 + m;
      memry[((size_t)bb * SS + ss) * HH + n] = (acc[nt][i] + bv[n]) * topw[rr];
    }
  }
}

// ------------------------------------------------------------------
// Kernel 5: attention. score_s = (Wk^T q)·mem_s + q·bk (factorized),
// softmax over S=64, read = attn @ memory. 1 block/batch.
// ------------------------------------------------------------------
__global__ void attn_kernel(const int* __restrict__ seq,
                            const float* __restrict__ embed,
                            const float* __restrict__ Wq,
                            const float* __restrict__ bq,
                            const float* __restrict__ Wk,
                            const float* __restrict__ bk,
                            const float* __restrict__ memry,
                            float* __restrict__ readb) {
  __shared__ float q[HH];
  __shared__ float qk[HH];
  __shared__ float sc[SS];
  __shared__ float red[256];
  const int b = blockIdx.x;
  const float* erow = embed + (size_t)seq[b * LL + (LL - 2)] * HH;

  for (int j = threadIdx.x; j < HH; j += 256) {
    const float* w = Wq + (size_t)j * HH;
    float s = 0.f;
    for (int k = 0; k < HH; ++k) s += erow[k] * w[k];
    q[j] = s + bq[j];
  }
  __syncthreads();

  for (int k = threadIdx.x; k < HH; k += 256) {
    float s = 0.f;
    for (int j = 0; j < HH; ++j) s += q[j] * Wk[(size_t)j * HH + k];
    qk[k] = s;
  }
  float part = 0.f;
  for (int j = threadIdx.x; j < HH; j += 256) part += q[j] * bk[j];
  red[threadIdx.x] = part;
  __syncthreads();
  for (int o = 128; o > 0; o >>= 1) {
    if (threadIdx.x < o) red[threadIdx.x] += red[threadIdx.x + o];
    __syncthreads();
  }
  float qbk = red[0];
  __syncthreads();

  if (threadIdx.x < SS) {
    const float* ms = memry + ((size_t)b * SS + threadIdx.x) * HH;
    float s = 0.f;
    for (int k = 0; k < HH; ++k) s += qk[k] * ms[k];
    sc[threadIdx.x] = (s + qbk) * (1.0f / 32.0f);   // /sqrt(H)
  }
  __syncthreads();

  float mx = -1e30f;
  for (int s2 = 0; s2 < SS; ++s2) mx = fmaxf(mx, sc[s2]);
  float sum = 0.f;
  for (int s2 = 0; s2 < SS; ++s2) sum += __expf(sc[s2] - mx);

  for (int k = threadIdx.x; k < HH; k += 256) {
    float r = 0.f;
    for (int s2 = 0; s2 < SS; ++s2)
      r += (__expf(sc[s2] - mx) / sum) * memry[((size_t)b * SS + s2) * HH + k];
    readb[b * HH + k] = r;
  }
}

// ------------------------------------------------------------------
// Kernel 6: logits = read @ Wo.T + bo. M=32 (2 tiles), N=50257 (3142
// tiles, guarded tail). Block 128 (4 waves), each wave one N-tile.
// ------------------------------------------------------------------
__global__ void logits_gemm_kernel(const float* __restrict__ readb,
                                   const float* __restrict__ Wo,
                                   const float* __restrict__ bo,
                                   float* __restrict__ out) {
  const int wave = threadIdx.x >> 5, lane = threadIdx.x & 31;
  const int ntile = blockIdx.x * 4 + wave;
  if (ntile >= 3142) return;
  const int m = lane & 15, hi = lane >> 4;
  const int col = ntile * 16 + m;

  v8f acc0 = v8f_zero(), acc1 = v8f_zero();
  for (int k0 = 0; k0 < HH; k0 += 32) {
    v16h bf;
    if (col < VV) {
      bf = load_bT_f32(Wo + (size_t)col * HH, k0, hi);
    } else {
#pragma unroll
      for (int i = 0; i < 16; ++i) bf[i] = (_Float16)0.f;
    }
    v16h a0 = load_a_f32(readb + (size_t)m * HH, k0, hi);
    v16h a1 = load_a_f32(readb + (size_t)(16 + m) * HH, k0, hi);
    acc0 = __builtin_amdgcn_wmma_f32_16x16x32_f16(false, a0, false, bf,
                                                  (short)0, acc0, false, false);
    acc1 = __builtin_amdgcn_wmma_f32_16x16x32_f16(false, a1, false, bf,
                                                  (short)0, acc1, false, false);
  }
#pragma unroll
  for (int i = 0; i < 8; ++i) {
    int n = ntile * 16 + m;
    if (n < VV) {
      int r0 = i + hi*8;
      out[(size_t)r0 * VV + n]        = acc0[i] + bo[n];
      out[(size_t)(16 + r0) * VV + n] = acc1[i] + bo[n];
    }
  }
}

// ------------------------------------------------------------------
extern "C" void kernel_launch(void* const* d_in, const int* in_sizes, int n_in,
                              void* d_out, int out_size, void* d_ws, size_t ws_size,
                              hipStream_t stream) {
  (void)in_sizes; (void)n_in; (void)out_size; (void)ws_size;
  const int*   seq = (const int*)  d_in[0];
  const float* u   = (const float*)d_in[1];
  const float* emb = (const float*)d_in[2];
  const float* Wih = (const float*)d_in[3];
  const float* Whh = (const float*)d_in[4];
  const float* bih = (const float*)d_in[5];
  const float* bhh = (const float*)d_in[6];
  const float* Wg  = (const float*)d_in[7];
  const float* bg  = (const float*)d_in[8];
  const float* Wv  = (const float*)d_in[9];
  const float* bv  = (const float*)d_in[10];
  const float* Wq  = (const float*)d_in[11];
  const float* bq  = (const float*)d_in[12];
  const float* Wk  = (const float*)d_in[13];
  const float* bk  = (const float*)d_in[14];
  const float* Wo  = (const float*)d_in[15];
  const float* bo  = (const float*)d_in[16];
  float* out = (float*)d_out;

  char* ws = (char*)d_ws;
  size_t off = 0;
  auto wsalloc = [&](size_t bytes) -> void* {
    void* p = ws + off;
    off += (bytes + 255) & ~(size_t)255;
    return p;
  };
  _Float16* xg      = (_Float16*)wsalloc((size_t)BB * LL * GD * 2);    // 256 MB
  _Float16* embed16 = (_Float16*)wsalloc((size_t)VV * HH * 2);         // 103 MB
  _Float16* wih16   = (_Float16*)wsalloc((size_t)GD * HH * 2);         // 4 MB
  _Float16* whh16   = (_Float16*)wsalloc((size_t)GD * HALFD * 2);      // 2 MB
  float* h     = (float*)wsalloc((size_t)BB * HALFD * 4);
  float* c     = (float*)wsalloc((size_t)BB * HALFD * 4);
  float* gbuf  = (float*)wsalloc((size_t)BB * GD * 4);
  float* gate  = (float*)wsalloc((size_t)BB * LL * 4);
  int*   tidx  = (int*)  wsalloc((size_t)BB * KK * 4);
  float* tw    = (float*)wsalloc((size_t)BB * KK * 4);
  float* memry = (float*)wsalloc((size_t)BB * SS * HH * 4);            // 8 MB
  float* readb = (float*)wsalloc((size_t)BB * HH * 4);
  unsigned* bar = (unsigned*)wsalloc(256);

  hipMemsetAsync(h, 0, (size_t)BB * HALFD * 4, stream);
  hipMemsetAsync(c, 0, (size_t)BB * HALFD * 4, stream);
  hipMemsetAsync(memry, 0, (size_t)BB * SS * HH * 4, stream);
  hipMemsetAsync(bar, 0, 256, stream);

  cvt_f16_kernel<<<(GD * HALFD + 255) / 256, 256, 0, stream>>>(Whh, whh16, GD * HALFD);
  cvt_f16_kernel<<<(GD * HH + 255) / 256, 256, 0, stream>>>(Wih, wih16, GD * HH);
  cvt_f16_kernel<<<(VV * HH + 255) / 256, 256, 0, stream>>>(emb, embed16, VV * HH);

  xg_gemm_kernel<<<dim3((BB * LL) / 128, GD / 128), 256, 0, stream>>>(seq, embed16, wih16, bih, bhh, xg);
  lstm_scan_kernel<<<SCAN_BLOCKS, 128, 0, stream>>>(xg, whh16, Wg, bg, h, c, gbuf, gate, bar);
  topk_kernel<<<BB, 256, 0, stream>>>(gate, u, tidx, tw);
  mem_gemm_kernel<<<dim3((BB * KK) / 16, HH / 256), 128, 0, stream>>>(seq, emb, Wv, bv, tidx, tw, memry);
  attn_kernel<<<BB, 256, 0, stream>>>(seq, emb, Wq, bq, Wk, bk, memry, readb);
  logits_gemm_kernel<<<786, 128, 0, stream>>>(readb, Wo, bo, out);
}
